// SimpleMoELayer_11003706212956
// MI455X (gfx1250) — matrix operand order repaired and checked by reading, not compiled, not run on hardware
//
#include <hip/hip_runtime.h>
#include <hip/hip_bf16.h>
#include <math.h>
#include <stdint.h>

typedef __attribute__((ext_vector_type(16))) __bf16 v16bf;
typedef __attribute__((ext_vector_type(8)))  float  v8f;

#define HDIM 1024
#define DDIM 2048
#define NEXP 8
#define NTOK 2048      // B*S
#define CAP  2048      // per-expert token list capacity

// ---------------------------------------------------------------------------
// x (fp32) -> bf16
// ---------------------------------------------------------------------------
__global__ void cvt_x_kernel(const float* __restrict__ x, __bf16* __restrict__ xb) {
  int i = (blockIdx.x * 256 + threadIdx.x) * 4;
  float4 v = *(const float4*)(x + i);
  union { __bf16 h[4]; uint2 u; } pk;
  pk.h[0] = (__bf16)v.x; pk.h[1] = (__bf16)v.y;
  pk.h[2] = (__bf16)v.z; pk.h[3] = (__bf16)v.w;
  *(uint2*)(xb + i) = pk.u;
}

// ---------------------------------------------------------------------------
// W1/W2 (fp32, row-major) -> bf16 WMMA B-fragment layout.
// Fragment = 32(K) x 16(N) tile stored as [lane][16 bf16]:
//   lane n   (n<16): elem j <-> B[k0+(j&7)+(j>>3)*16  ][n]   (k0 = kb*32)
//   lane n+16      : elem j <-> B[k0+8+(j&7)+(j>>3)*16][n]
// One wave per fragment; LDS transpose for coalesced source reads.
// W1 frags: [e][kb:H/32=32][nt:D/16=128]   W2 frags: [e][kb:D/32=64][nt:H/16=64]
// ---------------------------------------------------------------------------
__global__ void cvt_w_kernel(const float* __restrict__ W1, const float* __restrict__ W2,
                             __bf16* __restrict__ W1s, __bf16* __restrict__ W2s) {
  __shared__ float tsm[8][32][17];   // per-wave 32x16 tile (+pad)
  int w = threadIdx.x >> 5, lane = threadIdx.x & 31;
  int f = blockIdx.x * 8 + w;

  const float* src;
  __bf16* dst;
  int stride;
  if (f < NEXP * 32 * 128) {                       // W1 fragment
    int e = f / (32 * 128); int rem = f % (32 * 128);
    int kb = rem / 128;     int nt = rem % 128;
    src = W1 + (size_t)e * HDIM * DDIM + (size_t)(kb * 32) * DDIM + nt * 16;
    stride = DDIM;
    dst = W1s + (size_t)f * 512;
  } else {                                          // W2 fragment
    int f2 = f - NEXP * 32 * 128;
    int e = f2 / (64 * 64); int rem = f2 % (64 * 64);
    int kb = rem / 64;      int nt = rem % 64;
    src = W2 + (size_t)e * DDIM * HDIM + (size_t)(kb * 32) * HDIM + nt * 16;
    stride = HDIM;
    dst = W2s + (size_t)f2 * 512;
  }

  // lane loads source row k = lane (16 contiguous floats)
  const float4* s4 = (const float4*)(src + (size_t)lane * stride);
  float4 r0 = s4[0], r1 = s4[1], r2 = s4[2], r3 = s4[3];
  float* row = tsm[w][lane];
  row[0]=r0.x; row[1]=r0.y; row[2]=r0.z; row[3]=r0.w;
  row[4]=r1.x; row[5]=r1.y; row[6]=r1.z; row[7]=r1.w;
  row[8]=r2.x; row[9]=r2.y; row[10]=r2.z; row[11]=r2.w;
  row[12]=r3.x; row[13]=r3.y; row[14]=r3.z; row[15]=r3.w;
  __syncthreads();

  int cn = lane & 15, kh = lane >> 4;
  union { __bf16 h[16]; uint4 q[2]; } pk;
  #pragma unroll
  for (int j = 0; j < 16; ++j) pk.h[j] = (__bf16)tsm[w][kh * 8 + (j & 7) + (j >> 3) * 16][cn];
  *(uint4*)(dst + lane * 16)     = pk.q[0];
  *(uint4*)(dst + lane * 16 + 8) = pk.q[1];
}

// ---------------------------------------------------------------------------
// Router: one wave per token. logits = x @ Wr + br; softmax; top-2;
// scatter (token*2+slot, prob) into per-expert lists.
// ---------------------------------------------------------------------------
__global__ void moe_router_kernel(const float* __restrict__ x, const float* __restrict__ Wr,
                                  const float* __restrict__ br,
                                  int* __restrict__ cnt, int* __restrict__ toklist,
                                  float* __restrict__ problist) {
  int wave = threadIdx.x >> 5;
  int lane = threadIdx.x & 31;
  int t = blockIdx.x * 8 + wave;

  float acc[NEXP];
  #pragma unroll
  for (int e = 0; e < NEXP; ++e) acc[e] = 0.f;

  const float* xr = x + (size_t)t * HDIM;
  for (int h = lane; h < HDIM; h += 32) {
    float xv = xr[h];
    const float4* wr = (const float4*)(Wr + (size_t)h * NEXP);
    float4 w0 = wr[0], w1 = wr[1];
    acc[0] += xv * w0.x; acc[1] += xv * w0.y; acc[2] += xv * w0.z; acc[3] += xv * w0.w;
    acc[4] += xv * w1.x; acc[5] += xv * w1.y; acc[6] += xv * w1.z; acc[7] += xv * w1.w;
  }
  #pragma unroll
  for (int off = 16; off > 0; off >>= 1) {
    #pragma unroll
    for (int e = 0; e < NEXP; ++e) acc[e] += __shfl_xor(acc[e], off, 32);
  }

  if (lane == 0) {
    float lg[NEXP], m = -1e30f;
    #pragma unroll
    for (int e = 0; e < NEXP; ++e) { lg[e] = acc[e] + br[e]; m = fmaxf(m, lg[e]); }
    float s = 0.f;
    #pragma unroll
    for (int e = 0; e < NEXP; ++e) { lg[e] = __expf(lg[e] - m); s += lg[e]; }
    float inv = 1.f / s;
    int i1 = 0; float p1 = -1.f;
    #pragma unroll
    for (int e = 0; e < NEXP; ++e) { float pe = lg[e] * inv; if (pe > p1) { p1 = pe; i1 = e; } }
    int i2 = 0; float p2 = -1.f;
    #pragma unroll
    for (int e = 0; e < NEXP; ++e) {
      float pe = lg[e] * inv;
      if (e != i1 && pe > p2) { p2 = pe; i2 = e; }
    }
    int pos1 = atomicAdd(&cnt[i1], 1);
    toklist[i1 * CAP + pos1] = t * 2 + 0; problist[i1 * CAP + pos1] = p1;
    int pos2 = atomicAdd(&cnt[i2], 1);
    toklist[i2 * CAP + pos2] = t * 2 + 1; problist[i2 * CAP + pos2] = p2;
  }
}

// ---------------------------------------------------------------------------
// Per-expert tile GEMM, split over D-halves for 2x parallelism.
// Block = 8 waves, blockIdx.y = e*2 + half. Each block:
//   h_half = relu(xtile @ W1[e][:, half*1024:+1024] + b1_half)   (16 x 1024)
//   y_part = h_half @ W2[e][half*1024:+1024, :] (+ b2 if half==1) (16 x 1024)
//   slot[k][half][token,:] = p * y_part
// ---------------------------------------------------------------------------
__global__ void __launch_bounds__(256, 1) moe_expert_gemm_kernel(
    const __bf16* __restrict__ W1s, const __bf16* __restrict__ W2s,
    const __bf16* __restrict__ xb,
    const float* __restrict__ b1, const float* __restrict__ b2,
    const int* __restrict__ cnt, const int* __restrict__ toklist,
    const float* __restrict__ problist,
    float* __restrict__ s00, float* __restrict__ s01,
    float* __restrict__ s10, float* __restrict__ s11) {
  __shared__ __bf16 xtile[16 * 1024];  // gathered token rows (bf16)
  __shared__ __bf16 hbuf[16 * 1024];   // relu(h) for this D-half (bf16)

  int ey = blockIdx.y;
  int e = ey >> 1;
  int dh = ey & 1;
  int n = cnt[e];
  int tiles = (n + 15) >> 4;
  int bx = blockIdx.x;
  if (bx >= tiles) return;

  int tid = threadIdx.x;
  int wave = tid >> 5, lane = tid & 31;
  int cl = lane & 15, kh = lane >> 4;

  // per-lane metadata for its 8 rows (rows kh*8 .. kh*8+7)
  int myTok[8]; int myK[8]; float myP[8];
  #pragma unroll
  for (int v = 0; v < 8; ++v) {
    int g = bx * 16 + kh * 8 + v;
    if (g < n) {
      int tv = toklist[e * CAP + g];
      myTok[v] = tv >> 1; myK[v] = tv & 1; myP[v] = problist[e * CAP + g];
    } else { myTok[v] = -1; myK[v] = 0; myP[v] = 0.f; }
  }

  // cooperative gather: 16 rows x 1024 bf16 into LDS via async copy
  {
    int row  = tid >> 4;
    int colb = (tid & 15) * 64;
    int g = bx * 16 + row;
    int tv = (g < n) ? toklist[e * CAP + g] : -1;
    __bf16* dp = xtile + row * HDIM + colb;
    if (tv >= 0) {
      const __bf16* sp = xb + (size_t)(tv >> 1) * HDIM + colb;
      unsigned ldsoff = (unsigned)(uintptr_t)dp;
      unsigned long long ga = (unsigned long long)(uintptr_t)sp;
      #pragma unroll
      for (int j = 0; j < 8; ++j) {
        asm volatile("global_load_async_to_lds_b128 %0, %1, off"
                     :: "v"(ldsoff + (unsigned)(j * 16)),
                        "v"(ga + (unsigned long long)(j * 16))
                     : "memory");
      }
    } else {
      uint4 z = make_uint4(0, 0, 0, 0);
      uint4* d4 = (uint4*)dp;
      #pragma unroll
      for (int j = 0; j < 8; ++j) d4[j] = z;
    }
    asm volatile("s_wait_asynccnt 0" ::: "memory");
  }
  __syncthreads();

  v8f acc[8];
  #pragma unroll
  for (int t = 0; t < 8; ++t) {
    #pragma unroll
    for (int j = 0; j < 8; ++j) acc[t][j] = 0.f;
  }

  // ----- GEMM1: h cols dh*1024 + wave*128 + t*16, K over H -----
  for (int kb = 0; kb < 32; ++kb) {
    union { uint4 q[2]; v16bf v; } A;
    const __bf16* ap = xtile + cl * HDIM + kb * 32 + kh * 8;
    A.q[0] = *(const uint4*)(ap);
    A.q[1] = *(const uint4*)(ap + 16);
    const __bf16* bp = W1s +
        (((size_t)(e * 32 + kb)) * 128 + (size_t)(dh * 64 + wave * 8)) * 512 + lane * 16;
    __builtin_prefetch(bp + (size_t)128 * 512, 0, 1);  // next kb
    v16bf Bf[8];
    #pragma unroll
    for (int t = 0; t < 8; ++t) Bf[t] = *(const v16bf*)(bp + (size_t)t * 512);
    #pragma unroll
    for (int t = 0; t < 8; ++t) {
      acc[t] = __builtin_amdgcn_wmma_f32_16x16x32_bf16(
          false, A.v, false, Bf[t], (short)0, acc[t], false, false);
    }
  }

  // epilogue: + b1, relu, -> bf16 hbuf
  #pragma unroll
  for (int t = 0; t < 8; ++t) {
    int cloc = wave * 128 + t * 16 + cl;
    float bb = b1[(size_t)e * DDIM + dh * 1024 + cloc];
    #pragma unroll
    for (int v = 0; v < 8; ++v) {
      float hv = acc[t][v] + bb;
      hv = hv > 0.f ? hv : 0.f;
      hbuf[(v + kh * 8) * 1024 + cloc] = (__bf16)hv;
    }
  }
  __syncthreads();

  // ----- GEMM2: y cols wave*128 + t*16, K over this half's 1024 -----
  #pragma unroll
  for (int t = 0; t < 8; ++t) {
    #pragma unroll
    for (int j = 0; j < 8; ++j) acc[t][j] = 0.f;
  }
  for (int ks = 0; ks < 32; ++ks) {
    union { uint4 q[2]; v16bf v; } A;
    const __bf16* ap = hbuf + cl * 1024 + ks * 32 + kh * 8;
    A.q[0] = *(const uint4*)(ap);
    A.q[1] = *(const uint4*)(ap + 16);
    int kb2 = dh * 32 + ks;
    const __bf16* bp = W2s +
        (((size_t)(e * 64 + kb2)) * 64 + (size_t)(wave * 8)) * 512 + lane * 16;
    __builtin_prefetch(bp + (size_t)64 * 512, 0, 1);  // next ks
    v16bf Bf[8];
    #pragma unroll
    for (int t = 0; t < 8; ++t) Bf[t] = *(const v16bf*)(bp + (size_t)t * 512);
    #pragma unroll
    for (int t = 0; t < 8; ++t) {
      acc[t] = __builtin_amdgcn_wmma_f32_16x16x32_bf16(
          false, A.v, false, Bf[t], (short)0, acc[t], false, false);
    }
  }

  // final epilogue: (+ b2 on half 1), scale by prob, scatter to slot buffer
  float* d0 = dh ? s01 : s00;
  float* d1 = dh ? s11 : s10;
  #pragma unroll
  for (int t = 0; t < 8; ++t) {
    int c = wave * 128 + t * 16 + cl;
    float bb = dh ? b2[(size_t)e * HDIM + c] : 0.f;
    #pragma unroll
    for (int v = 0; v < 8; ++v) {
      if (myTok[v] >= 0) {
        float y = (acc[t][v] + bb) * myP[v];
        float* dst = myK[v] ? d1 : d0;
        dst[(size_t)myTok[v] * HDIM + c] = y;
      }
    }
  }
}

// ---------------------------------------------------------------------------
// out = s00 + s01 + s10 + s11
// ---------------------------------------------------------------------------
__global__ void combine_out_kernel(const float* __restrict__ a,
                                   const float* __restrict__ b,
                                   const float* __restrict__ c,
                                   const float* __restrict__ d,
                                   float* __restrict__ out) {
  int i = (blockIdx.x * 256 + threadIdx.x) * 4;
  float4 va = *(const float4*)(a + i);
  float4 vb = *(const float4*)(b + i);
  float4 vc = *(const float4*)(c + i);
  float4 vd = *(const float4*)(d + i);
  float4 o = make_float4(va.x + vb.x + vc.x + vd.x,
                         va.y + vb.y + vc.y + vd.y,
                         va.z + vb.z + vc.z + vd.z,
                         va.w + vb.w + vc.w + vd.w);
  *(float4*)(out + i) = o;
}

// ---------------------------------------------------------------------------
extern "C" void kernel_launch(void* const* d_in, const int* in_sizes, int n_in,
                              void* d_out, int out_size, void* d_ws, size_t ws_size,
                              hipStream_t stream) {
  const float* x  = (const float*)d_in[0];
  const float* Wr = (const float*)d_in[1];
  const float* br = (const float*)d_in[2];
  const float* W1 = (const float*)d_in[3];
  const float* b1 = (const float*)d_in[4];
  const float* W2 = (const float*)d_in[5];
  const float* b2 = (const float*)d_in[6];
  float* out = (float*)d_out;

  char* ws = (char*)d_ws;
  size_t off = 0;
  __bf16* W1s = (__bf16*)(ws + off); off += (size_t)NEXP * HDIM * DDIM * 2;   // 32 MB
  __bf16* W2s = (__bf16*)(ws + off); off += (size_t)NEXP * DDIM * HDIM * 2;   // 32 MB
  __bf16* xb  = (__bf16*)(ws + off); off += (size_t)NTOK * HDIM * 2;          // 4 MB
  float*  s00 = (float*)(ws + off);  off += (size_t)NTOK * HDIM * 4;          // 8 MB
  float*  s01 = (float*)(ws + off);  off += (size_t)NTOK * HDIM * 4;          // 8 MB
  float*  s10 = (float*)(ws + off);  off += (size_t)NTOK * HDIM * 4;          // 8 MB
  float*  s11 = (float*)(ws + off);  off += (size_t)NTOK * HDIM * 4;          // 8 MB
  int*    cnt = (int*)(ws + off);    off += 256;
  int*    tok = (int*)(ws + off);    off += (size_t)NEXP * CAP * 4;
  float*  prl = (float*)(ws + off);  off += (size_t)NEXP * CAP * 4;

  hipMemsetAsync(cnt, 0, 256, stream);

  cvt_x_kernel<<<2048, 256, 0, stream>>>(x, xb);                 // 2M elems / (256*4)
  cvt_w_kernel<<<8192, 256, 0, stream>>>(W1, W2, W1s, W2s);      // 65536 frags / 8 waves
  moe_router_kernel<<<256, 256, 0, stream>>>(x, Wr, br, cnt, tok, prl);
  moe_expert_gemm_kernel<<<dim3(128, NEXP * 2), 256, 0, stream>>>(
      W1s, W2s, xb, b1, b2, cnt, tok, prl, s00, s01, s10, s11);
  combine_out_kernel<<<2048, 256, 0, stream>>>(s00, s01, s10, s11, out);
}
